// ASFormerLayer_88098369176204
// MI455X (gfx1250) — matrix operand order repaired and checked by reading, not compiled, not run on hardware
//
#include <hip/hip_runtime.h>
#include <hip/hip_bf16.h>

// ---------------------------------------------------------------------------
// ASFormer layer on gfx1250 (MI455X).
//  - All GEMMs: v_wmma_f32_16x16x32_bf16, f32 accumulate.
//  - B tiles staged in LDS via gfx1250 async copy (global_load_async_to_lds,
//    ASYNCcnt) with double buffering; A fragments software-pipelined across a
//    raw split barrier (no fence -> no loadcnt drain).
//  - Flash-style causal attention, 1 wave per 16-query tile.
//  - Fused bias/relu/bf16-requant/LN epilogues.
// ---------------------------------------------------------------------------

typedef __bf16 v8bf  __attribute__((ext_vector_type(8)));
typedef __bf16 v16bf __attribute__((ext_vector_type(16)));
typedef float  v8f   __attribute__((ext_vector_type(8)));

#define DEVI __device__ __forceinline__

static constexpr int BB  = 2;
static constexpr int T   = 2048;
static constexpr int C   = 512;
static constexpr int H   = 8;
static constexpr int HD  = 64;
static constexpr int C3  = 3 * C;      // 1536
static constexpr int KW  = 3;          // conv kernel width

// LDS B-tile: 64 rows (n) x 32 k bf16, row stride padded to 80B
static constexpr int BROW = 80;        // bytes per LDS row (64B data + 16B pad)

// ---------------- WMMA fragment helpers (contiguous k order, same in A and B) -----

DEVI v16bf frag_cat(v8bf lo, v8bf hi) {
  return __builtin_shufflevector(lo, hi, 0,1,2,3,4,5,6,7,8,9,10,11,12,13,14,15);
}

// A (16x32 bf16): lane holds row (lane&15), contiguous k-range k0..k0+15,
// k0=(lane>>4)*16
DEVI v16bf load_frag_a(const __bf16* base, long lda, int lane) {
  int m = lane & 15, k0 = (lane >> 4) * 16;
  const __bf16* p = base + (long)m * lda;
  v8bf lo = *reinterpret_cast<const v8bf*>(p + k0);
  v8bf hi = *reinterpret_cast<const v8bf*>(p + 8 + k0);
  return frag_cat(lo, hi);
}

// B (32x16 bf16) from transposed storage Bt[N][K]: lane holds col (lane&15),
// contiguous k-range k0=(lane>>4)*16
DEVI v16bf load_frag_b(const __bf16* baseT, long ldb, int lane) {
  int n = lane & 15, k0 = (lane >> 4) * 16;
  const __bf16* p = baseT + (long)n * ldb + k0;
  v8bf lo = *reinterpret_cast<const v8bf*>(p);
  v8bf hi = *reinterpret_cast<const v8bf*>(p + 8);
  return frag_cat(lo, hi);
}

// B fragment from the padded LDS tile (subtile t: rows t*16..t*16+15)
DEVI v16bf load_frag_b_lds(const char* buf, int t, int lane) {
  int n = lane & 15, k0 = (lane >> 4) * 16;
  const __bf16* p =
      reinterpret_cast<const __bf16*>(buf + (t * 16 + n) * BROW) + k0;
  v8bf lo = *reinterpret_cast<const v8bf*>(p);
  v8bf hi = *reinterpret_cast<const v8bf*>(p + 8);
  return frag_cat(lo, hi);
}

DEVI v8f wmma_bf16(v16bf a, v16bf b, v8f c) {
  return __builtin_amdgcn_wmma_f32_16x16x32_bf16(false, a, false, b, (short)0, c,
                                                 false, false);
}

// gfx1250 async copy: LDS[laddr .. +15] = MEM[gaddr .. +15]  (ASYNCcnt)
DEVI void async_copy_b128(const void* gaddr, void* lptr) {
  unsigned laddr = (unsigned)(unsigned long long)
      (__attribute__((address_space(3))) char*)lptr;
  asm volatile("global_load_async_to_lds_b128 %0, %1, off"
               :: "v"(laddr), "v"(gaddr) : "memory");
}
DEVI void wait_async0() {
  asm volatile("s_wait_asynccnt 0x0" ::: "memory");
}
// split workgroup barrier WITHOUT the fence of __syncthreads(): no vmem
// drain, so software-pipelined global loads stay in flight across it.
DEVI void wg_barrier() {
  asm volatile("s_barrier_signal -1\n\ts_barrier_wait -1" ::: "memory");
}
// compiler-only ordering fence (same-wave LDS ops are in-order at the LDS)
DEVI void compiler_fence() { asm volatile("" ::: "memory"); }

// ---------------- bf16 WMMA GEMM:  C = A(M,K) * Bt(N,K)^T + bias -----------
// 4 waves / WG; WG tile 128x64 (wave: 32x64, 8 WMMAs per 32-k step).
// B tile (64x32) double-buffered in LDS, filled by async DMA.
// epi: 1=write f32, 2=write bf16, 4=relu.  Batched via gridDim.z.

__global__ __launch_bounds__(128) void gemm_bf16_wmma(
    const __bf16* __restrict__ A, long lda, long strideA,
    const __bf16* __restrict__ Bt, long ldb,
    const float* __restrict__ bias,
    float* __restrict__ Cf, __bf16* __restrict__ Cb, long ldc, long strideC,
    int M, int N, int Kd, int epi)
{
  const int wave = threadIdx.x >> 5;
  const int lane = threadIdx.x & 31;
  const long bz = blockIdx.z;
  const int m0 = blockIdx.x * 128 + wave * 32;   // wave owns rows m0..m0+31
  const int n0 = blockIdx.y * 64;

  __shared__ __align__(16) char bshm[2][64 * BROW];

  const __bf16* Ab = A + bz * strideA;
  const __bf16* Arow0 = Ab + (long)m0 * lda;
  const __bf16* Arow1 = Ab + (long)(m0 + 16) * lda;

  // async-fill one 64x32 B tile (4KB): 256 x 16B chunks over 128 threads
  auto issue_copy = [&](int kk0, int buf) {
#pragma unroll
    for (int c = 0; c < 2; ++c) {
      int lin = threadIdx.x + c * 128;     // chunk index
      int row = lin >> 2, k16 = lin & 3;   // 4 chunks per 64B row
      const __bf16* g = Bt + (long)(n0 + row) * ldb + kk0 + k16 * 8;
      async_copy_b128(g, &bshm[buf][row * BROW + k16 * 16]);
    }
  };

  const v8f zero = {0.f,0.f,0.f,0.f,0.f,0.f,0.f,0.f};
  v8f acc0[4], acc1[4];
#pragma unroll
  for (int t = 0; t < 4; ++t) { acc0[t] = zero; acc1[t] = zero; }

  // prologue: first B tile + first A fragments
  issue_copy(0, 0);
  v16bf a0 = load_frag_a(Arow0, lda, lane);
  v16bf a1 = load_frag_a(Arow1, lda, lane);
  wait_async0();
  wg_barrier();

  int cur = 0;
  for (int k0 = 0; k0 < Kd; k0 += 32) {
    const int nxt = k0 + 32;
    if (nxt < Kd) issue_copy(nxt, cur ^ 1);   // DMA next tile while computing

    v16bf a0u = a0, a1u = a1;
    if (nxt < Kd) {                           // prefetch next A fragments
      a0 = load_frag_a(Arow0 + nxt, lda, lane);
      a1 = load_frag_a(Arow1 + nxt, lda, lane);
    }
    // batch all B-fragment LDS reads, then the 8 WMMAs
    v16bf bfr[4];
#pragma unroll
    for (int t = 0; t < 4; ++t) bfr[t] = load_frag_b_lds(bshm[cur], t, lane);
#pragma unroll
    for (int t = 0; t < 4; ++t) acc0[t] = wmma_bf16(a0u, bfr[t], acc0[t]);
#pragma unroll
    for (int t = 0; t < 4; ++t) acc1[t] = wmma_bf16(a1u, bfr[t], acc1[t]);

    wait_async0();       // next tile landed
    wg_barrier();        // everyone done reading cur / sees cur^1
    cur ^= 1;
  }

  const int nlo = lane & 15, hi = lane >> 4;
#pragma unroll
  for (int s = 0; s < 2; ++s) {
    v8f* acc = s ? acc1 : acc0;
#pragma unroll
    for (int t = 0; t < 4; ++t) {
#pragma unroll
      for (int r = 0; r < 8; ++r) {
        const long row = m0 + s * 16 + r + 8 * hi;
        const long col = n0 + t * 16 + nlo;
        float v = acc[t][r] + bias[col];
        if (epi & 4) v = fmaxf(v, 0.0f);
        if (epi & 1) Cf[bz * strideC + row * ldc + col] = v;
        if (epi & 2) Cb[bz * strideC + row * ldc + col] = (__bf16)v;
      }
    }
  }
  (void)M; (void)N;
}

// ---------------- flash-style causal attention -----------------------------
// 1 wave per 16-query tile of one (b,h). Scores Q@K^T via WMMA; P goes
// D-layout -> LDS -> A-layout; P@V uses vT[b,h,d,T].

__global__ __launch_bounds__(32) void attn_flash_wmma(
    const __bf16* __restrict__ qkv,   // [B,T,3C] bf16
    const __bf16* __restrict__ vT,    // [B,H,HD,T] bf16
    __bf16* __restrict__ out)         // [B,T,C] bf16
{
  const int lane = threadIdx.x;
  const int qi = blockIdx.x;          // query tile (T/16)
  const int h  = blockIdx.y;
  const int b  = blockIdx.z;
  const int qbase = qi * 16;
  const int lo = lane & 15, hi = lane >> 4;

  const __bf16* Qbase = qkv + ((long)(b * T + qbase)) * C3 + h * HD;
  const __bf16* Kbase = qkv + (long)b * T * C3 + C + h * HD;
  const __bf16* Vtb   = vT + ((long)(b * H + h) * HD) * T;

  v16bf aQ[2];
#pragma unroll
  for (int c = 0; c < 2; ++c) aQ[c] = load_frag_a(Qbase + c * 32, C3, lane);

  const v8f zero = {0.f,0.f,0.f,0.f,0.f,0.f,0.f,0.f};
  v8f o[4];
#pragma unroll
  for (int t = 0; t < 4; ++t) o[t] = zero;
  float rm[8], rs[8];
#pragma unroll
  for (int r = 0; r < 8; ++r) { rm[r] = -1e30f; rs[r] = 0.0f; }

  __shared__ __align__(16) __bf16 ldsP[16 * 32];

  for (int jj = 0; jj <= qi; jj += 2) {
    float s[2][8];
#pragma unroll
    for (int t2 = 0; t2 < 2; ++t2) {
      const int jt = jj + t2;
      if (jt <= qi) {
        const __bf16* Kt = Kbase + (long)jt * 16 * C3;
        v8f sc = zero;
#pragma unroll
        for (int c = 0; c < 2; ++c) {
          v16bf bK = load_frag_b(Kt + c * 32, C3, lane);
          sc = wmma_bf16(aQ[c], bK, sc);
        }
#pragma unroll
        for (int r = 0; r < 8; ++r) {
          float v = sc[r] * 0.125f;                 // 1/sqrt(64)
          if (jt * 16 + lo > qbase + r + 8 * hi) v = -1e30f;  // causal mask
          s[t2][r] = v;
        }
      } else {
#pragma unroll
        for (int r = 0; r < 8; ++r) s[t2][r] = -1e30f;
      }
    }

    // online softmax (row lives in 16 lanes of one half-wave)
    float nm[8], al[8];
#pragma unroll
    for (int r = 0; r < 8; ++r) {
      float v = fmaxf(s[0][r], s[1][r]);
      v = fmaxf(v, __shfl_xor(v, 1));
      v = fmaxf(v, __shfl_xor(v, 2));
      v = fmaxf(v, __shfl_xor(v, 4));
      v = fmaxf(v, __shfl_xor(v, 8));
      nm[r] = fmaxf(rm[r], v);
      al[r] = __expf(rm[r] - nm[r]);
      rm[r] = nm[r];
    }
#pragma unroll
    for (int t = 0; t < 4; ++t)
#pragma unroll
      for (int r = 0; r < 8; ++r) o[t][r] *= al[r];

#pragma unroll
    for (int r = 0; r < 8; ++r) {
      float ps = 0.0f;
#pragma unroll
      for (int t2 = 0; t2 < 2; ++t2) {
        float p = __expf(s[t2][r] - nm[r]);
        ps += p;
        ldsP[(r + 8 * hi) * 32 + t2 * 16 + lo] = (__bf16)p;
      }
      ps += __shfl_xor(ps, 1);
      ps += __shfl_xor(ps, 2);
      ps += __shfl_xor(ps, 4);
      ps += __shfl_xor(ps, 8);
      rs[r] = rs[r] * al[r] + ps;
    }
    compiler_fence();  // same-wave LDS ops are in-order; just stop reordering

    // reload P as A-fragment (16x32)
    v16bf aP;
    {
      const __bf16* p = ldsP + lo * 32;
      int k0 = hi * 16;
      v8bf l8 = *reinterpret_cast<const v8bf*>(p + k0);
      v8bf h8 = *reinterpret_cast<const v8bf*>(p + 8 + k0);
      aP = frag_cat(l8, h8);
    }
#pragma unroll
    for (int t = 0; t < 4; ++t) {
      const __bf16* vp = Vtb + (long)(t * 16 + lo) * T + jj * 16 + hi * 16;
      v8bf l8 = *reinterpret_cast<const v8bf*>(vp);
      v8bf h8 = *reinterpret_cast<const v8bf*>(vp + 8);
      o[t] = wmma_bf16(aP, frag_cat(l8, h8), o[t]);
    }
    compiler_fence();
  }

  float inv[8];
#pragma unroll
  for (int r = 0; r < 8; ++r) inv[r] = 1.0f / rs[r];
#pragma unroll
  for (int t = 0; t < 4; ++t)
#pragma unroll
    for (int r = 0; r < 8; ++r)
      out[((long)(b * T + qbase + r + 8 * hi)) * C + h * HD + t * 16 + lo] =
          (__bf16)(o[t][r] * inv[r]);
}

// ---------------- fused residual-add + LayerNorm ---------------------------

__global__ __launch_bounds__(128) void add_ln_kernel(
    const float* __restrict__ x, const float* __restrict__ r,
    const float* __restrict__ g, const float* __restrict__ be,
    float* __restrict__ outf, __bf16* __restrict__ outb)
{
  const long row = blockIdx.x;
  const int tid = threadIdx.x;
  const long base = row * C;
  float v[4], s = 0.f, s2 = 0.f;
#pragma unroll
  for (int i = 0; i < 4; ++i) {
    int c = tid + i * 128;
    v[i] = x[base + c] + r[base + c];
    s += v[i];
    s2 += v[i] * v[i];
  }
  __shared__ float red[2][4];
#pragma unroll
  for (int m = 16; m >= 1; m >>= 1) {
    s  += __shfl_xor(s, m);
    s2 += __shfl_xor(s2, m);
  }
  const int w = tid >> 5;
  if ((tid & 31) == 0) { red[0][w] = s; red[1][w] = s2; }
  __syncthreads();
  float S = red[0][0] + red[0][1] + red[0][2] + red[0][3];
  float S2 = red[1][0] + red[1][1] + red[1][2] + red[1][3];
  const float mean = S / C;
  const float var = S2 / C - mean * mean;
  const float rstd = rsqrtf(var + 1e-5f);
#pragma unroll
  for (int i = 0; i < 4; ++i) {
    int c = tid + i * 128;
    float y = (v[i] - mean) * rstd * g[c] + be[c];
    outf[base + c] = y;
    outb[base + c] = (__bf16)y;
  }
}

// ---------------- small data-movement / conversion kernels -----------------

__global__ void k_pad_x(const float* __restrict__ x, __bf16* __restrict__ xp) {
  long idx = (long)blockIdx.x * blockDim.x + threadIdx.x;
  const long total = (long)BB * (T + 2) * C;
  if (idx >= total) return;
  int i = idx % C;
  long tp = (idx / C) % (T + 2);
  long b = idx / ((long)C * (T + 2));
  xp[idx] = (tp < 2) ? (__bf16)0.0f : (__bf16)x[(b * T + (tp - 2)) * C + i];
}

// Wt[n][k] = W[k][n], f32 -> bf16
__global__ void k_transpose_w(const float* __restrict__ W, __bf16* __restrict__ Wt,
                              int Kd, int N) {
  long idx = (long)blockIdx.x * blockDim.x + threadIdx.x;
  if (idx >= (long)Kd * N) return;
  int k = idx % Kd;
  int n = idx / Kd;
  Wt[(long)n * Kd + k] = (__bf16)W[(long)k * N + n];
}

// conv OIH [o,i,k] -> Wt[o][k*C+i]
__global__ void k_conv_w(const float* __restrict__ W, __bf16* __restrict__ Wt) {
  long idx = (long)blockIdx.x * blockDim.x + threadIdx.x;
  if (idx >= (long)C * C * KW) return;
  int kk = idx % (C * KW);
  int o = idx / (C * KW);
  int k = kk / C, i = kk % C;
  Wt[idx] = (__bf16)W[((long)o * C + i) * KW + k];
}

// vT[b,h,d,t] = qkv[b,t, 2C + h*HD + d]
__global__ void k_build_vT(const __bf16* __restrict__ qkv, __bf16* __restrict__ vT) {
  long idx = (long)blockIdx.x * blockDim.x + threadIdx.x;
  const long total = (long)BB * H * HD * T;
  if (idx >= total) return;
  int t = idx % T;
  int d = (idx / T) % HD;
  int h = (idx / ((long)T * HD)) % H;
  int b = idx / ((long)T * HD * H);
  vT[idx] = qkv[((long)(b * T + t)) * C3 + 2 * C + h * HD + d];
}

// ---------------------------------------------------------------------------

static inline size_t alignup(size_t x) { return (x + 255) & ~(size_t)255; }

extern "C" void kernel_launch(void* const* d_in, const int* in_sizes, int n_in,
                              void* d_out, int out_size, void* d_ws, size_t ws_size,
                              hipStream_t stream) {
  const float* x      = (const float*)d_in[0];
  const float* conv_w = (const float*)d_in[1];
  const float* conv_b = (const float*)d_in[2];
  const float* g1     = (const float*)d_in[3];
  const float* b1     = (const float*)d_in[4];
  const float* qkv_w  = (const float*)d_in[5];
  const float* qkv_b  = (const float*)d_in[6];
  const float* proj_w = (const float*)d_in[7];
  const float* proj_b = (const float*)d_in[8];
  const float* g2     = (const float*)d_in[9];
  const float* b2     = (const float*)d_in[10];
  const float* ffn_w1 = (const float*)d_in[11];
  const float* ffn_b1 = (const float*)d_in[12];
  const float* ffn_w2 = (const float*)d_in[13];
  const float* ffn_b2 = (const float*)d_in[14];
  const float* g3     = (const float*)d_in[15];
  const float* b3     = (const float*)d_in[16];
  float* out = (float*)d_out;

  const long M = (long)BB * T;          // 4096 rows

  // ---- workspace carve-up ----
  char* w = (char*)d_ws;
  size_t off = 0;
  auto take = [&](size_t bytes) { char* p = w + off; off += alignup(bytes); return p; };
  __bf16* xpad   = (__bf16*)take((size_t)BB * (T + 2) * C * 2);
  __bf16* wconvT = (__bf16*)take((size_t)C * C3 * 2);        // [512][1536]
  __bf16* wqkvT  = (__bf16*)take((size_t)C3 * C * 2);        // [1536][512]
  __bf16* wprojT = (__bf16*)take((size_t)C * C * 2);         // [512][512]
  __bf16* wffn1T = (__bf16*)take((size_t)2 * C * C * 2);     // [1024][512]
  __bf16* wffn2T = (__bf16*)take((size_t)C * 2 * C * 2);     // [512][1024]
  float*  gemmf  = (float*)take((size_t)M * C * 4);          // conv/proj/ffn2 out
  float*  x1f    = (float*)take((size_t)M * C * 4);
  __bf16* x1b    = (__bf16*)take((size_t)M * C * 2);
  __bf16* qkvb   = (__bf16*)take((size_t)M * C3 * 2);
  __bf16* vTb    = (__bf16*)take((size_t)BB * H * HD * T * 2);
  __bf16* attnb  = (__bf16*)take((size_t)M * C * 2);
  float*  x2f    = (float*)take((size_t)M * C * 4);
  __bf16* x2b    = (__bf16*)take((size_t)M * C * 2);
  __bf16* hb     = (__bf16*)take((size_t)M * 2 * C * 2);     // ffn hidden
  (void)ws_size; (void)in_sizes; (void)n_in; (void)out_size;

  // ---- weight prep (bf16, transposed) ----
  {
    long n;
    n = (long)C * C3;
    k_conv_w<<<(n + 255) / 256, 256, 0, stream>>>(conv_w, wconvT);
    n = (long)C * C3;
    k_transpose_w<<<(n + 255) / 256, 256, 0, stream>>>(qkv_w, wqkvT, C, C3);
    n = (long)C * C;
    k_transpose_w<<<(n + 255) / 256, 256, 0, stream>>>(proj_w, wprojT, C, C);
    n = (long)C * 2 * C;
    k_transpose_w<<<(n + 255) / 256, 256, 0, stream>>>(ffn_w1, wffn1T, C, 2 * C);
    n = (long)C * 2 * C;
    k_transpose_w<<<(n + 255) / 256, 256, 0, stream>>>(ffn_w2, wffn2T, 2 * C, C);
    n = (long)BB * (T + 2) * C;
    k_pad_x<<<(n + 255) / 256, 256, 0, stream>>>(x, xpad);
  }

  // ---- 1) causal conv as GEMM over padded rows (lda = C, Kd = 3C) ----
  gemm_bf16_wmma<<<dim3(T / 128, C / 64, BB), 128, 0, stream>>>(
      xpad, C, (long)(T + 2) * C, wconvT, C3, conv_b,
      gemmf, nullptr, C, (long)T * C, T, C, C3, /*epi=*/1);

  // LN1: x1 = LN(x + conv)
  add_ln_kernel<<<M, 128, 0, stream>>>(x, gemmf, g1, b1, x1f, x1b);

  // ---- 2) QKV projection -> bf16 [B,T,3C] ----
  gemm_bf16_wmma<<<dim3(M / 128, C3 / 64, 1), 128, 0, stream>>>(
      x1b, C, 0, wqkvT, C, qkv_b,
      nullptr, qkvb, C3, 0, M, C3, C, /*epi=*/2);

  // build V^T [B,H,HD,T]
  {
    long n = (long)BB * H * HD * T;
    k_build_vT<<<(n + 255) / 256, 256, 0, stream>>>(qkvb, vTb);
  }

  // ---- 3) causal attention ----
  attn_flash_wmma<<<dim3(T / 16, H, BB), 32, 0, stream>>>(qkvb, vTb, attnb);

  // ---- 4) output projection ----
  gemm_bf16_wmma<<<dim3(M / 128, C / 64, 1), 128, 0, stream>>>(
      attnb, C, 0, wprojT, C, proj_b,
      gemmf, nullptr, C, 0, M, C, C, /*epi=*/1);

  // LN2: x2 = LN(x1 + proj)
  add_ln_kernel<<<M, 128, 0, stream>>>(x1f, gemmf, g2, b2, x2f, x2b);

  // ---- 5) FFN ----
  gemm_bf16_wmma<<<dim3(M / 128, (2 * C) / 64, 1), 128, 0, stream>>>(
      x2b, C, 0, wffn1T, C, ffn_b1,
      nullptr, hb, 2 * C, 0, M, 2 * C, C, /*epi=*/2 | 4);   // relu -> bf16

  gemm_bf16_wmma<<<dim3(M / 128, C / 64, 1), 128, 0, stream>>>(
      hb, 2 * C, 0, wffn2T, 2 * C, ffn_b2,
      gemmf, nullptr, C, 0, M, C, 2 * C, /*epi=*/1);

  // LN3: out = LN(x2 + ffn)  (bf16 copy discarded into x1b)
  add_ln_kernel<<<M, 128, 0, stream>>>(x2f, gemmf, g3, b3, out, x1b);
}
